// TruthGPTDistanceAttentionBlock_8555574854218
// MI455X (gfx1250) — matrix operand order repaired and checked by reading, not compiled, not run on hardware
//
#include <hip/hip_runtime.h>
#include <hip/hip_fp16.h>
#include <hip/hip_bf16.h>

#define HDIM 768
#define NH 12
#define HD 64
#define FF 3072
#define SEQ 1024
#define BAT 2
#define BH (BAT*NH)
#define MROWS (BAT*SEQ)

typedef _Float16 half_t;
typedef __attribute__((ext_vector_type(16))) _Float16 v16h;
typedef __attribute__((ext_vector_type(8)))  _Float16 v8h;
typedef __attribute__((ext_vector_type(8)))  float    v8f;
typedef __attribute__((ext_vector_type(4)))  int      v4i;

// ---------------------------------------------------------------------------
// Async global->LDS helpers (CDNA5 GLOBAL_LOAD_ASYNC_TO_LDS_B128, ASYNCcnt).
// Builtin signature on this toolchain (from the round-2 diagnostic):
//   (v4i addrspace(1)* gsrc, v4i addrspace(3)* ldst, imm offset, imm cpol)
// LDS generic addresses keep the LDS byte offset in the low 32 bits, so the
// uintptr_t round-trip into an addrspace(3) pointer is the correct operand.
// ---------------------------------------------------------------------------
__device__ __forceinline__ void async_copy16(const void* g, void* l) {
#if __has_builtin(__builtin_amdgcn_global_load_async_to_lds_b128)
  __builtin_amdgcn_global_load_async_to_lds_b128(
      (__attribute__((address_space(1))) v4i*)(uintptr_t)g,
      (__attribute__((address_space(3))) v4i*)(uintptr_t)l, 0, 0);
#else
  *(uint4*)l = *(const uint4*)g;
#endif
}
__device__ __forceinline__ void async_wait_all() {
#if __has_builtin(__builtin_amdgcn_global_load_async_to_lds_b128)
#if __has_builtin(__builtin_amdgcn_s_wait_asynccnt)
  __builtin_amdgcn_s_wait_asynccnt(0);
#else
  asm volatile("s_wait_asynccnt 0" ::: "memory");
#endif
#endif
}

// ---------------------------------------------------------------------------
// Weight convert+transpose: W[K][N] f32 -> Wt[N][K] f16 (coalesced writes)
// ---------------------------------------------------------------------------
__global__ __launch_bounds__(256) void transpose_w_kernel(
    const float* __restrict__ W, half_t* __restrict__ Wt, int K, int N) {
  long i = (long)blockIdx.x * blockDim.x + threadIdx.x;
  long total = (long)K * N;
  if (i >= total) return;
  long n = i / K, k = i % K;
  Wt[i] = (half_t)W[k * (long)N + n];
}

// ---------------------------------------------------------------------------
// LayerNorm: one wave (32 lanes) per 768-wide row, f32 in -> f16 out
// ---------------------------------------------------------------------------
__global__ __launch_bounds__(256) void ln_kernel(
    const float* __restrict__ x, const float* __restrict__ g,
    const float* __restrict__ b, half_t* __restrict__ out, int rows) {
  int wid = threadIdx.x >> 5, lane = threadIdx.x & 31;
  int row = blockIdx.x * (blockDim.x >> 5) + wid;
  if (row >= rows) return;
  const float* xr = x + (size_t)row * HDIM;
  float vals[24];
  float s = 0.f;
#pragma unroll
  for (int i = 0; i < 24; i++) { vals[i] = xr[lane + i * 32]; s += vals[i]; }
#pragma unroll
  for (int m = 16; m; m >>= 1) s += __shfl_xor(s, m, 32);
  float mu = s * (1.0f / HDIM);
  float v = 0.f;
#pragma unroll
  for (int i = 0; i < 24; i++) { float d = vals[i] - mu; v += d * d; }
#pragma unroll
  for (int m = 16; m; m >>= 1) v += __shfl_xor(v, m, 32);
  float inv = rsqrtf(v * (1.0f / HDIM) + 1e-5f);
  half_t* orow = out + (size_t)row * HDIM;
#pragma unroll
  for (int i = 0; i < 24; i++) {
    int c = lane + i * 32;
    orow[c] = (half_t)((vals[i] - mu) * inv * g[c] + b[c]);
  }
}

// ---------------------------------------------------------------------------
// V[b,s, h*64+d] (f16, [MROWS x H]) -> Vt[((b*NH+h)*HD+d)*S + s]
// ---------------------------------------------------------------------------
__global__ __launch_bounds__(256) void transpose_v_kernel(
    const half_t* __restrict__ Vh, half_t* __restrict__ Vt) {
  long i = (long)blockIdx.x * blockDim.x + threadIdx.x;  // output index
  int s = (int)(i % SEQ);
  int d = (int)((i / SEQ) % HD);
  int bh = (int)(i / ((long)SEQ * HD));
  int b = bh / NH, h = bh % NH;
  Vt[i] = Vh[((long)(b * SEQ + s)) * HDIM + h * HD + d];
}

// ---------------------------------------------------------------------------
// Register-blocked WMMA GEMM: C[MxN] = A[MxK] f16 x Bt[NxK]^T f16 + bias.
// One wave owns a 64x32 macro-tile: 4 M-tiles x 2 N-tiles = 8 f32 accumulators
// (64 VGPRs). Per 32-deep k-step: 4 A-frags + 2 B-frags -> 8 independent
// v_wmma_f32_16x16x32_f16 (no D->A/B hazards, deep WMMA pipelining).
// MODE 0: f16 out | 1: f16 + exact GELU | 2: f32 + residual | 3: f16, per-head
//         context layout (z = b*NH+h, column base b*S*H + h*HD)
// ---------------------------------------------------------------------------
#define MT 4
#define NT 2
template <int MODE>
__global__ __launch_bounds__(256) void gemm_wmma(
    const half_t* __restrict__ A, const half_t* __restrict__ Bt,
    const float* __restrict__ bias, void* __restrict__ Cout,
    const float* __restrict__ resid, int M, int N, int K, int ldc,
    long aBatch, long btBatch, long cBatch) {
  int wid = threadIdx.x >> 5, lane = threadIdx.x & 31;
  int nmt = N >> 5;                       // macro-tiles along N (32 wide)
  int t = blockIdx.x * (blockDim.x >> 5) + wid;
  if (t >= (M >> 6) * nmt) return;        // macro-tiles along M (64 tall)
  int mt = t / nmt, nt = t % nmt;
  int z = blockIdx.y;
  const half_t* Az = A + (long)z * aBatch;
  const half_t* Bz = Bt + (long)z * btBatch;

  int r = lane & 15, hs = lane >> 4;
  const half_t* abase = Az + (size_t)(mt * 64 + r) * K;
  const half_t* bbase = Bz + (size_t)(nt * 32 + r) * K;

  v8f acc[MT][NT];
#pragma unroll
  for (int mi = 0; mi < MT; mi++)
#pragma unroll
    for (int ni = 0; ni < NT; ni++)
      acc[mi][ni] = (v8f){0.f, 0.f, 0.f, 0.f, 0.f, 0.f, 0.f, 0.f};

  for (int k0 = 0; k0 < K; k0 += 32) {
    v16h a[MT], bf[NT];
#pragma unroll
    for (int mi = 0; mi < MT; mi++) {
      const half_t* ap = abase + (size_t)mi * 16 * K + k0;
      a[mi] = *(const v16h*)(ap + 16 * hs);
    }
#pragma unroll
    for (int ni = 0; ni < NT; ni++)
      bf[ni] = *(const v16h*)(bbase + (size_t)ni * 16 * K + k0 + 16 * hs);
#pragma unroll
    for (int mi = 0; mi < MT; mi++)
#pragma unroll
      for (int ni = 0; ni < NT; ni++)
        acc[mi][ni] = __builtin_amdgcn_wmma_f32_16x16x32_f16(
            false, a[mi], false, bf[ni], (short)0, acc[mi][ni], false, false);
  }

  size_t cbase;
  if (MODE == 3)
    cbase = (size_t)(z / NH) * SEQ * HDIM + (size_t)(z % NH) * HD;
  else
    cbase = (size_t)z * (size_t)cBatch;
#pragma unroll
  for (int mi = 0; mi < MT; mi++) {
#pragma unroll
    for (int ni = 0; ni < NT; ni++) {
      int col = nt * 32 + ni * 16 + r;
      float bvv = bias ? bias[col] : 0.f;
#pragma unroll
      for (int j = 0; j < 8; j++) {
        int m = mt * 64 + mi * 16 + j + 8 * hs;
        float c = acc[mi][ni][j] + bvv;
        size_t idx = cbase + (size_t)m * ldc + col;
        if (MODE == 0 || MODE == 3) {
          ((half_t*)Cout)[idx] = (half_t)c;
        } else if (MODE == 1) {
          ((half_t*)Cout)[idx] =
              (half_t)(0.5f * c * (1.0f + erff(c * 0.70710678f)));
        } else {
          ((float*)Cout)[idx] = c + resid[idx];
        }
      }
    }
  }
}

// ---------------------------------------------------------------------------
// L1 distance + softmax. Block: 256 threads (8 waves), each wave owns one q
// row; K head tile staged in LDS (128 rows x 64 f16, padded rows) via the
// CDNA5 async global->LDS path. Packed-f16 abs-diff accumulation; wave32
// shfl softmax. Writes attn f32 (d_out) and attn f16 (A for attn@V WMMA).
// ---------------------------------------------------------------------------
#define KCHUNK 128
__global__ __launch_bounds__(256) void l1_attn_kernel(
    const half_t* __restrict__ Qh, const half_t* __restrict__ Kh,
    const float* __restrict__ lamPtr, float* __restrict__ attnF,
    half_t* __restrict__ attnH) {
  __shared__ __align__(16) half_t Ks[KCHUNK][72];
  int wid = threadIdx.x >> 5, lane = threadIdx.x & 31;
  int bh = blockIdx.y, b = bh / NH, h = bh % NH;
  int q = blockIdx.x * 8 + wid;
  float sc = -lamPtr[0] * 0.125f;  // 1/sqrt(HD) = 0.125

  const __half2* qrow =
      (const __half2*)(Qh + ((size_t)(b * SEQ + q)) * HDIM + h * HD);
  __half2 qv[32];
#pragma unroll
  for (int i = 0; i < 32; i++) qv[i] = qrow[i];

  const half_t* Kbase = Kh + ((size_t)b * SEQ) * HDIM + h * HD;
  float scores[32];

  for (int kc = 0; kc < SEQ / KCHUNK; kc++) {
    __syncthreads();
    // async stage: 128 rows x 128B = 1024 x 16B; 4 async b128 copies/thread
#pragma unroll
    for (int i = 0; i < 4; i++) {
      int idx = threadIdx.x + i * 256;
      int row = idx >> 3, seg = idx & 7;
      async_copy16(Kbase + (size_t)(kc * KCHUNK + row) * HDIM + seg * 8,
                   &Ks[row][seg * 8]);
    }
    async_wait_all();
    __syncthreads();
#pragma unroll
    for (int i = 0; i < 4; i++) {
      int kk = i * 32 + lane;
      const __half2* kr = (const __half2*)(&Ks[kk][0]);
      __half2 acc2 = __float2half2_rn(0.f);
#pragma unroll
      for (int j = 0; j < 32; j++)
        acc2 = __hadd2(acc2, __habs2(__hsub2(qv[j], kr[j])));
      scores[kc * 4 + i] = sc * (__low2float(acc2) + __high2float(acc2));
    }
  }

  float mx = -1e30f;
#pragma unroll
  for (int i = 0; i < 32; i++) mx = fmaxf(mx, scores[i]);
#pragma unroll
  for (int m = 16; m; m >>= 1) mx = fmaxf(mx, __shfl_xor(mx, m, 32));
  float sum = 0.f;
#pragma unroll
  for (int i = 0; i < 32; i++) { scores[i] = __expf(scores[i] - mx); sum += scores[i]; }
#pragma unroll
  for (int m = 16; m; m >>= 1) sum += __shfl_xor(sum, m, 32);
  float inv = 1.f / sum;

  float*  arow = attnF + ((size_t)bh * SEQ + q) * SEQ;
  half_t* hrow = attnH + ((size_t)bh * SEQ + q) * SEQ;
#pragma unroll
  for (int i = 0; i < 32; i++) {
    int k = (i >> 2) * KCHUNK + (i & 3) * 32 + lane;
    float p = scores[i] * inv;
    arow[k] = p;
    hrow[k] = (half_t)p;
  }
}

// ---------------------------------------------------------------------------
// Workspace layout (bytes)
// ---------------------------------------------------------------------------
static constexpr size_t SZ_HH  = (size_t)HDIM * HDIM * 2;   // f16 768x768
static constexpr size_t SZ_HFF = (size_t)HDIM * FF * 2;     // f16 768x3072
static constexpr size_t SZ_MH  = (size_t)MROWS * HDIM * 2;  // f16 2048x768
static constexpr size_t SZ_ATT = (size_t)BH * SEQ * SEQ * 2;
static constexpr size_t O_WQT = 0;
static constexpr size_t O_WKT = O_WQT + SZ_HH;
static constexpr size_t O_WVT = O_WKT + SZ_HH;
static constexpr size_t O_WOT = O_WVT + SZ_HH;
static constexpr size_t O_W1T = O_WOT + SZ_HH;
static constexpr size_t O_W2T = O_W1T + SZ_HFF;
static constexpr size_t O_XLN = O_W2T + SZ_HFF;
static constexpr size_t O_QH  = O_XLN + SZ_MH;
static constexpr size_t O_KH  = O_QH + SZ_MH;
static constexpr size_t O_VH  = O_KH + SZ_MH;
static constexpr size_t O_VT  = O_VH + SZ_MH;
static constexpr size_t O_AH  = O_VT + SZ_MH;
static constexpr size_t O_CTX = O_AH + SZ_ATT;
static constexpr size_t O_O1  = O_CTX + SZ_MH;              // f32 2048x768
static constexpr size_t O_Y2  = O_O1 + (size_t)MROWS * HDIM * 4;
static constexpr size_t O_FF1 = O_Y2 + SZ_MH;

extern "C" void kernel_launch(void* const* d_in, const int* in_sizes, int n_in,
                              void* d_out, int out_size, void* d_ws,
                              size_t ws_size, hipStream_t stream) {
  const float* hs   = (const float*)d_in[0];
  const float* ln1g = (const float*)d_in[1];
  const float* ln1b = (const float*)d_in[2];
  const float* Wq = (const float*)d_in[3];
  const float* bq = (const float*)d_in[4];
  const float* Wk = (const float*)d_in[5];
  const float* bk = (const float*)d_in[6];
  const float* Wv = (const float*)d_in[7];
  const float* bv = (const float*)d_in[8];
  const float* Wo = (const float*)d_in[9];
  const float* bo = (const float*)d_in[10];
  const float* lam  = (const float*)d_in[11];
  const float* ln2g = (const float*)d_in[12];
  const float* ln2b = (const float*)d_in[13];
  const float* W1 = (const float*)d_in[14];
  const float* b1 = (const float*)d_in[15];
  const float* W2 = (const float*)d_in[16];
  const float* b2 = (const float*)d_in[17];

  char* w = (char*)d_ws;
  half_t* WqT = (half_t*)(w + O_WQT);
  half_t* WkT = (half_t*)(w + O_WKT);
  half_t* WvT = (half_t*)(w + O_WVT);
  half_t* WoT = (half_t*)(w + O_WOT);
  half_t* W1T = (half_t*)(w + O_W1T);
  half_t* W2T = (half_t*)(w + O_W2T);
  half_t* xln = (half_t*)(w + O_XLN);
  half_t* Qh  = (half_t*)(w + O_QH);
  half_t* Kh  = (half_t*)(w + O_KH);
  half_t* Vh  = (half_t*)(w + O_VH);
  half_t* Vt  = (half_t*)(w + O_VT);
  half_t* aH  = (half_t*)(w + O_AH);
  half_t* ctx = (half_t*)(w + O_CTX);
  float*  o1  = (float*)(w + O_O1);
  half_t* y2  = (half_t*)(w + O_Y2);
  half_t* f1  = (half_t*)(w + O_FF1);

  float* outF  = (float*)d_out;
  float* attnF = outF + (size_t)BAT * SEQ * HDIM;

  // 1) weights -> f16 transposed [N x K]
  transpose_w_kernel<<<(HDIM * HDIM) / 256, 256, 0, stream>>>(Wq, WqT, HDIM, HDIM);
  transpose_w_kernel<<<(HDIM * HDIM) / 256, 256, 0, stream>>>(Wk, WkT, HDIM, HDIM);
  transpose_w_kernel<<<(HDIM * HDIM) / 256, 256, 0, stream>>>(Wv, WvT, HDIM, HDIM);
  transpose_w_kernel<<<(HDIM * HDIM) / 256, 256, 0, stream>>>(Wo, WoT, HDIM, HDIM);
  transpose_w_kernel<<<(HDIM * FF) / 256, 256, 0, stream>>>(W1, W1T, HDIM, FF);
  transpose_w_kernel<<<(HDIM * FF) / 256, 256, 0, stream>>>(W2, W2T, FF, HDIM);

  // 2) LN1 -> f16
  ln_kernel<<<MROWS / 8, 256, 0, stream>>>(hs, ln1g, ln1b, xln, MROWS);

  // 3) Q,K,V projections (WMMA f16, 64x32 macro-tiles)
  int tQKV = (MROWS / 64) * (HDIM / 32);  // 768 macro-tiles
  gemm_wmma<0><<<dim3(tQKV / 8, 1), 256, 0, stream>>>(
      xln, WqT, bq, Qh, nullptr, MROWS, HDIM, HDIM, HDIM, 0, 0, 0);
  gemm_wmma<0><<<dim3(tQKV / 8, 1), 256, 0, stream>>>(
      xln, WkT, bk, Kh, nullptr, MROWS, HDIM, HDIM, HDIM, 0, 0, 0);
  gemm_wmma<0><<<dim3(tQKV / 8, 1), 256, 0, stream>>>(
      xln, WvT, bv, Vh, nullptr, MROWS, HDIM, HDIM, HDIM, 0, 0, 0);

  // 4) per-head V transpose -> [BH, HD, S]
  transpose_v_kernel<<<((long)BH * HD * SEQ) / 256, 256, 0, stream>>>(Vh, Vt);

  // 5) L1 distance + softmax -> attn f32 (d_out) + f16 staging
  l1_attn_kernel<<<dim3(SEQ / 8, BH), 256, 0, stream>>>(Qh, Kh, lam, attnF, aH);

  // 6) attn @ V (batched WMMA, per-head epilogue into [B,S,H] context)
  int tAV = (SEQ / 64) * (HD / 32);  // 32 macro-tiles per head
  gemm_wmma<3><<<dim3(tAV / 8, BH), 256, 0, stream>>>(
      aH, Vt, nullptr, ctx, nullptr, SEQ, HD, SEQ, HDIM,
      (long)SEQ * SEQ, (long)HD * SEQ, 0);

  // 7) output projection + residual -> f32
  gemm_wmma<2><<<dim3(tQKV / 8, 1), 256, 0, stream>>>(
      ctx, WoT, bo, o1, hs, MROWS, HDIM, HDIM, HDIM, 0, 0, 0);

  // 8) LN2 -> f16
  ln_kernel<<<MROWS / 8, 256, 0, stream>>>(o1, ln2g, ln2b, y2, MROWS);

  // 9) FF1 + exact GELU -> f16
  int tFF1 = (MROWS / 64) * (FF / 32);  // 3072 macro-tiles
  gemm_wmma<1><<<dim3(tFF1 / 8, 1), 256, 0, stream>>>(
      y2, W1T, b1, f1, nullptr, MROWS, FF, HDIM, FF, 0, 0, 0);

  // 10) FF2 + residual -> final f32 output
  gemm_wmma<2><<<dim3(tQKV / 8, 1), 256, 0, stream>>>(
      f1, W2T, b2, outF, o1, MROWS, HDIM, FF, HDIM, 0, 0, 0);
}
